// SelfAttention_89644557403070
// MI455X (gfx1250) — matrix operand order repaired and checked by reading, not compiled, not run on hardware
//
#include <hip/hip_runtime.h>

// Self-attention (B=4, C=256, H=W=64, N=4096, CQ=32) for gfx1250 (MI455X).
// Flash-attention formulation, all GEMMs via v_wmma_f32_16x16x32_f16.
// v2: b128 tile loads in projections; software-pipelined S/PV with
//     double-buffered P in the attention kernel (1 barrier per key tile).

#define NDIM  4096
#define CDIM  256
#define BATCH 4

typedef __attribute__((ext_vector_type(16))) _Float16 v16h;
typedef __attribute__((ext_vector_type(4)))  _Float16 v4h;
typedef __attribute__((ext_vector_type(8)))  float    v8f;
typedef _Float16 half_t;

// ---------------------------------------------------------------------------
// WMMA fragment loaders (CDNA5 16x16x32 f16 layouts, wave32).
//
// A-matrix 16x32 (MxK): lane m=l&15; lanes 0-15 hold K {0-7,16-23},
// lanes 16-31 hold K {8-15,24-31}; VGPR j holds consecutive K pairs.
// => per lane: two contiguous 16B loads at half-dependent offsets.
static __device__ inline v16h load_a_frag(const half_t* p0, int stride_h) {
  const int lane = threadIdx.x & 31;
  const half_t* p = p0 + (lane & 15) * stride_h + ((lane >> 4) << 3);
  v16h r;
  ((float4*)&r)[0] = *(const float4*)(p);        // K khalf..khalf+7
  ((float4*)&r)[1] = *(const float4*)(p + 16);   // K 16+khalf..+7
  return r;
}

// B-matrix 32x16 (KxN): lane n=l&15; lanes 0-15 hold K {0-7,16-23},
// lanes 16-31 hold K {8-15,24-31} (same as A). Source array is row-major [col][k].
// => per lane: two contiguous 16B loads at half-dependent offsets.
static __device__ inline v16h load_b_frag(const half_t* p0, int stride_h) {
  const int lane = threadIdx.x & 31;
  const half_t* p = p0 + (lane & 15) * stride_h + ((lane >> 4) << 3);
  v16h r;
  ((float4*)&r)[0] = *(const float4*)(p);
  ((float4*)&r)[1] = *(const float4*)(p + 16);
  return r;
}

static __device__ inline v8f wmma_f16(v16h a, v16h b, v8f c) {
  return __builtin_amdgcn_wmma_f32_16x16x32_f16(false, a, false, b, (short)0, c,
                                                false, false);
}

// ---------------------------------------------------------------------------
// Kernel 1: qk projection.  out(n, 0:32)=q*(1/sqrt(C)), out(n, 32:64)=k.
// Per block: 64 n-rows x 64 cols, K-loop over C=256 in steps of 32.
__global__ __launch_bounds__(256) void proj_qk_kernel(
    const float* __restrict__ x, const float* __restrict__ Wq,
    const float* __restrict__ bq, const float* __restrict__ Wk,
    const float* __restrict__ bk, half_t* __restrict__ qk) {
  __shared__ half_t xT[64 * 40];  // [n][ch], padded stride 40 (16B aligned)
  __shared__ half_t WL[64 * 40];  // [outcol][ch]

  const int b   = blockIdx.x >> 6;
  const int n0  = (blockIdx.x & 63) * 64;
  const int tid = threadIdx.x;
  const int wid = tid >> 5;
  const int lane = tid & 31;
  const int rblk  = wid & 3;   // 16-row n strip
  const int cpair = wid >> 2;  // col tiles {cpair*32, cpair*32+16}

  v8f acc0 = {}, acc1 = {};
  for (int ch = 0; ch < CDIM; ch += 32) {
    __syncthreads();
    // x chunk (32ch x 64n) f32, b128 loads, transposed f16 into LDS
#pragma unroll
    for (int i = 0; i < 2; ++i) {
      int idx = i * 256 + tid;
      int c4 = idx & 15, r = idx >> 4;
      float4 f = *(const float4*)&x[(size_t)(b * CDIM + ch + r) * NDIM + n0 + c4 * 4];
      xT[(c4 * 4 + 0) * 40 + r] = (half_t)f.x;
      xT[(c4 * 4 + 1) * 40 + r] = (half_t)f.y;
      xT[(c4 * 4 + 2) * 40 + r] = (half_t)f.z;
      xT[(c4 * 4 + 3) * 40 + r] = (half_t)f.w;
    }
    // [Wq;Wk] chunk (64oc x 32ch), b128 loads, contiguous f16 rows
#pragma unroll
    for (int i = 0; i < 2; ++i) {
      int idx = i * 256 + tid;
      int r4 = idx & 7, oc = idx >> 3;
      const float* wsrc = (oc < 32) ? &Wq[oc * CDIM + ch + r4 * 4]
                                    : &Wk[(oc - 32) * CDIM + ch + r4 * 4];
      float4 f = *(const float4*)wsrc;
      v4h h = {(half_t)f.x, (half_t)f.y, (half_t)f.z, (half_t)f.w};
      *(v4h*)&WL[oc * 40 + r4 * 4] = h;
    }
    __syncthreads();
    v16h a  = load_a_frag(xT + rblk * 16 * 40, 40);
    v16h b0 = load_b_frag(WL + (cpair * 32 + 0) * 40, 40);
    v16h b1 = load_b_frag(WL + (cpair * 32 + 16) * 40, 40);
    acc0 = wmma_f16(a, b0, acc0);
    acc1 = wmma_f16(a, b1, acc1);
  }
  // epilogue: bias, fold 1/sqrt(256) into q half, store f16 (B,N,64)
  const int rbase = n0 + rblk * 16 + ((lane >> 4) << 3);
#pragma unroll
  for (int t = 0; t < 2; ++t) {
    v8f acc = t ? acc1 : acc0;
    int col = cpair * 32 + t * 16 + (lane & 15);
    float bias = (col < 32) ? bq[col] : bk[col - 32];
    float scl  = (col < 32) ? 0.0625f : 1.0f;
#pragma unroll
    for (int j = 0; j < 8; ++j) {
      qk[(size_t)(b * NDIM + rbase + j) * 64 + col] = (half_t)((acc[j] + bias) * scl);
    }
  }
}

// ---------------------------------------------------------------------------
// Kernel 2: v projection.  v(b,c,n) = Wv@x + bv, stored f16 (B,C,N).
// Per block: 64 c-rows x 64 n-cols.
__global__ __launch_bounds__(256) void proj_v_kernel(
    const float* __restrict__ x, const float* __restrict__ Wv,
    const float* __restrict__ bv, half_t* __restrict__ vout) {
  __shared__ half_t WL[64 * 40];  // [c][ch]
  __shared__ half_t xT[64 * 40];  // [n][ch]

  const int bid = blockIdx.x;
  const int b  = bid >> 8;
  const int c0 = ((bid >> 6) & 3) * 64;
  const int n0 = (bid & 63) * 64;
  const int tid = threadIdx.x, wid = tid >> 5, lane = tid & 31;
  const int rblk  = wid & 3;   // c strip
  const int npair = wid >> 2;  // n tiles {npair*32, npair*32+16}

  v8f acc0 = {}, acc1 = {};
  for (int ch = 0; ch < CDIM; ch += 32) {
    __syncthreads();
#pragma unroll
    for (int i = 0; i < 2; ++i) {
      int idx = i * 256 + tid;
      int r4 = idx & 7, cc = idx >> 3;
      float4 f = *(const float4*)&Wv[(c0 + cc) * CDIM + ch + r4 * 4];
      v4h h = {(half_t)f.x, (half_t)f.y, (half_t)f.z, (half_t)f.w};
      *(v4h*)&WL[cc * 40 + r4 * 4] = h;
    }
#pragma unroll
    for (int i = 0; i < 2; ++i) {
      int idx = i * 256 + tid;
      int c4 = idx & 15, r = idx >> 4;
      float4 f = *(const float4*)&x[(size_t)(b * CDIM + ch + r) * NDIM + n0 + c4 * 4];
      xT[(c4 * 4 + 0) * 40 + r] = (half_t)f.x;
      xT[(c4 * 4 + 1) * 40 + r] = (half_t)f.y;
      xT[(c4 * 4 + 2) * 40 + r] = (half_t)f.z;
      xT[(c4 * 4 + 3) * 40 + r] = (half_t)f.w;
    }
    __syncthreads();
    v16h a  = load_a_frag(WL + rblk * 16 * 40, 40);
    v16h b0 = load_b_frag(xT + (npair * 32 + 0) * 40, 40);
    v16h b1 = load_b_frag(xT + (npair * 32 + 16) * 40, 40);
    acc0 = wmma_f16(a, b0, acc0);
    acc1 = wmma_f16(a, b1, acc1);
  }
  const int chalf = (lane >> 4) << 3;
#pragma unroll
  for (int t = 0; t < 2; ++t) {
    v8f acc = t ? acc1 : acc0;
    int n = n0 + npair * 32 + t * 16 + (lane & 15);
#pragma unroll
    for (int j = 0; j < 8; ++j) {
      int c = c0 + rblk * 16 + chalf + j;
      vout[(size_t)(b * CDIM + c) * NDIM + n] = (half_t)(acc[j] + bv[c]);
    }
  }
}

// ---------------------------------------------------------------------------
// Kernel 3: flash attention + epilogue out = gamma*attn_out + x.
// Block = one batch b, one 64-query tile. 8 waves:
//   waves 0-3: S = Q K^T (one wmma per 16x16 S tile since d=32), online
//              softmax, write P(f16) + per-row alpha into the *other* LDS
//              buffer (software pipeline, double-buffered).
//   all waves: O^T(c,n) accumulation, O^T += V^T P^T  (A = v rows from
//              global/L2, B = P rows from LDS; both contiguous loads).
// One barrier per key tile.
__global__ __launch_bounds__(256) void attn_kernel(
    const half_t* __restrict__ qk, const half_t* __restrict__ vb,
    const float* __restrict__ x, const float* __restrict__ gamma,
    float* __restrict__ out) {
  __shared__ half_t Pb[2][64 * 72];  // P[n_local][m_local], stride 72
  __shared__ float alphaArr[2][64];
  __shared__ float linvArr[64];

  const int b  = blockIdx.x >> 6;
  const int q0 = (blockIdx.x & 63) * 64;
  const int tid = threadIdx.x, wid = tid >> 5, lane = tid & 31;
  const bool is_swave = wid < 4;
  const int qs = wid & 3;
  const float L2E = 1.44269504088896340736f;

  v8f acc[2][4] = {};  // O^T tiles: c strip = wid*32 (+cb*16), n tiles nb*16
  float mrow[8], lrow[8];
  v16h qfrag;
  if (is_swave) {
#pragma unroll
    for (int j = 0; j < 8; ++j) { mrow[j] = -3.0e38f; lrow[j] = 0.0f; }
    // Q rows (cols 0..31 of qk buffer); 1/sqrt(C) already folded in.
    qfrag = load_a_frag(qk + (size_t)(b * NDIM + q0 + qs * 16) * 64, 64);
  }

  // S + online softmax for key tile at kt, results into buffer `sel`.
  auto s_step = [&](int kt, int sel) {
    v8f s[4];
    const half_t* kbase = qk + (size_t)(b * NDIM + kt) * 64 + 32;  // K half
#pragma unroll
    for (int mb = 0; mb < 4; ++mb) {
      v16h bf = load_b_frag(kbase + mb * 16 * 64, 64);
      v8f z = {};
      s[mb] = wmma_f16(qfrag, bf, z);
    }
    const int rlocal = qs * 16 + ((lane >> 4) << 3);
#pragma unroll
    for (int j = 0; j < 8; ++j) {
      float rm = fmaxf(fmaxf(s[0][j], s[1][j]), fmaxf(s[2][j], s[3][j]));
      rm = fmaxf(rm, __shfl_xor(rm, 1));
      rm = fmaxf(rm, __shfl_xor(rm, 2));
      rm = fmaxf(rm, __shfl_xor(rm, 4));
      rm = fmaxf(rm, __shfl_xor(rm, 8));
      float mnew  = fmaxf(mrow[j], rm);
      float alpha = exp2f((mrow[j] - mnew) * L2E);
      mrow[j] = mnew;
      float psum = 0.0f;
#pragma unroll
      for (int mb = 0; mb < 4; ++mb) {
        float p = exp2f((s[mb][j] - mnew) * L2E);
        psum += p;
        Pb[sel][(rlocal + j) * 72 + mb * 16 + (lane & 15)] = (half_t)p;
      }
      psum += __shfl_xor(psum, 1);
      psum += __shfl_xor(psum, 2);
      psum += __shfl_xor(psum, 4);
      psum += __shfl_xor(psum, 8);
      lrow[j] = lrow[j] * alpha + psum;
      if ((lane & 15) == 0) alphaArr[sel][rlocal + j] = alpha;
    }
  };

  // prologue: produce P/alpha for tile 0 into buffer 0
  if (is_swave) s_step(0, 0);
  __syncthreads();

  for (int t = 0; t < NDIM / 64; ++t) {
    const int kt  = t * 64;
    const int buf = t & 1;
    // pipeline: S-waves produce tile t+1 into the other buffer
    if (is_swave && (t + 1) < NDIM / 64) s_step(kt + 64, buf ^ 1);

    // all waves: rescale O^T by alpha[n] (n = lane col), then O^T += V^T P^T
    float al[4];
#pragma unroll
    for (int nb = 0; nb < 4; ++nb) al[nb] = alphaArr[buf][nb * 16 + (lane & 15)];
#pragma unroll
    for (int cb = 0; cb < 2; ++cb)
#pragma unroll
      for (int nb = 0; nb < 4; ++nb)
#pragma unroll
        for (int j = 0; j < 8; ++j) acc[cb][nb][j] *= al[nb];

    const half_t* vrow = vb + (size_t)(b * CDIM + wid * 32) * NDIM + kt;
#pragma unroll
    for (int kb = 0; kb < 2; ++kb) {
      v16h a0 = load_a_frag(vrow + kb * 32, NDIM);              // c rows +0..15
      v16h a1 = load_a_frag(vrow + 16 * NDIM + kb * 32, NDIM);  // c rows +16..31
#pragma unroll
      for (int nb = 0; nb < 4; ++nb) {
        v16h bp = load_b_frag(Pb[buf] + nb * 16 * 72 + kb * 32, 72);
        acc[0][nb] = wmma_f16(a0, bp, acc[0][nb]);
        acc[1][nb] = wmma_f16(a1, bp, acc[1][nb]);
      }
    }
    // ensure: P[buf^1] writes visible for t+1; P[buf] reads done before t+1
    // overwrites it.
    __syncthreads();
  }

  if (is_swave) {
    const int rlocal = qs * 16 + ((lane >> 4) << 3);
    if ((lane & 15) == 0) {
#pragma unroll
      for (int j = 0; j < 8; ++j) linvArr[rlocal + j] = 1.0f / lrow[j];
    }
  }
  __syncthreads();

  const float g = gamma[0];
  float li[4];
#pragma unroll
  for (int nb = 0; nb < 4; ++nb) li[nb] = linvArr[nb * 16 + (lane & 15)];
#pragma unroll
  for (int cb = 0; cb < 2; ++cb) {
#pragma unroll
    for (int nb = 0; nb < 4; ++nb) {
      int n = q0 + nb * 16 + (lane & 15);
      int cbase = wid * 32 + cb * 16 + ((lane >> 4) << 3);
#pragma unroll
      for (int j = 0; j < 8; ++j) {
        size_t idx = (size_t)(b * CDIM + cbase + j) * NDIM + n;
        out[idx] = g * (acc[cb][nb][j] * li[nb]) + x[idx];
      }
    }
  }
}

// ---------------------------------------------------------------------------
extern "C" void kernel_launch(void* const* d_in, const int* in_sizes, int n_in,
                              void* d_out, int out_size, void* d_ws,
                              size_t ws_size, hipStream_t stream) {
  (void)in_sizes; (void)n_in; (void)out_size; (void)ws_size;
  const float* x     = (const float*)d_in[0];
  const float* Wq    = (const float*)d_in[1];
  const float* bq    = (const float*)d_in[2];
  const float* Wk    = (const float*)d_in[3];
  const float* bk    = (const float*)d_in[4];
  const float* Wv    = (const float*)d_in[5];
  const float* bv    = (const float*)d_in[6];
  const float* gamma = (const float*)d_in[7];
  float* out = (float*)d_out;

  half_t* qk = (half_t*)d_ws;                                           // 2 MB
  half_t* v  = (half_t*)((char*)d_ws + (size_t)BATCH * NDIM * 64 * 2);  // 8 MB

  proj_qk_kernel<<<BATCH * (NDIM / 64), 256, 0, stream>>>(x, Wq, bq, Wk, bk, qk);
  proj_v_kernel<<<BATCH * (CDIM / 64) * (NDIM / 64), 256, 0, stream>>>(x, Wv, bv, v);
  attn_kernel<<<BATCH * (NDIM / 64), 256, 0, stream>>>(qk, v, x, gamma, out);
}
